// LSTM_35510789604046
// MI455X (gfx1250) — matrix operand order repaired and checked by reading, not compiled, not run on hardware
//
#include <hip/hip_runtime.h>
#include <cstdint>

typedef __attribute__((ext_vector_type(16))) __bf16 v16bf;
typedef __attribute__((ext_vector_type(8)))  float  v8f;

#define T_STEPS 256
#define BATCH   16
#define B2      32
#define DIN     512
#define HID     1024
#define G4H     4096
#define OUTF    4096

// ---------- bf16 helpers (round-to-nearest-even) ----------
static __device__ __forceinline__ __bf16 f2bf(float f) {
  union { float f; unsigned u; } v; v.f = f;
  unsigned r = v.u + 0x7FFFu + ((v.u >> 16) & 1u);
  unsigned short h = (unsigned short)(r >> 16);
  return __builtin_bit_cast(__bf16, h);
}
static __device__ __forceinline__ float bf2f(__bf16 b) {
  unsigned short h = __builtin_bit_cast(unsigned short, b);
  union { unsigned u; float f; } v; v.u = ((unsigned)h) << 16;
  return v.f;
}

// ---------- WMMA fragment loaders ----------
// A fragment: 16x32 (MxK) bf16, row-major source with leading dim lda.
// lane: M = lane&15, half = lane>>4 selects K-offset 8; VGPR v -> K = (v>>2)*16 + (v&3)*2.
// Both K-groups are 16B-contiguous -> two b128 loads.
static __device__ __forceinline__ v16bf load_a_frag(const __bf16* A, int m0, int lda, int k0, int lane) {
  const __bf16* base = A + (size_t)(m0 + (lane & 15)) * lda + k0 + ((lane >> 4) << 3);
  v16bf f;
  uint4* fv = reinterpret_cast<uint4*>(&f);
  fv[0] = *reinterpret_cast<const uint4*>(base);
  fv[1] = *reinterpret_cast<const uint4*>(base + 16);
  return f;
}
// B fragment: 32x16 (KxN) bf16 fed from row-major W[N,K] (i.e. B = W^T).
// lane: N = lane&15, half = lane>>4 selects K-offset 8; VGPR v -> K = (v>>2)*16 + (v&3)*2 (same as A).
static __device__ __forceinline__ v16bf load_b_frag(const __bf16* W, int n0, int ldw, int k0, int lane) {
  const __bf16* base = W + (size_t)(n0 + (lane & 15)) * ldw + k0 + ((lane >> 4) << 3);
  v16bf f;
  uint4* fv = reinterpret_cast<uint4*>(&f);
  fv[0] = *reinterpret_cast<const uint4*>(base);
  fv[1] = *reinterpret_cast<const uint4*>(base + 16);
  return f;
}

// ---------- generic WMMA GEMM: out[M,N] = (A[M,K] * W[N,K]^T) * scale[n] + shift[n] ----------
// grid = (N/128, M/64), 256 threads = 8 waves; wave tile 16(M) x 64(N).
__global__ __launch_bounds__(256) void gemm_bf16_epi(
    const __bf16* __restrict__ A, const __bf16* __restrict__ W,
    const float* __restrict__ scale, const float* __restrict__ shift,
    __bf16* __restrict__ out_bf, float* __restrict__ out_f, int N, int K) {
  const int lane = threadIdx.x & 31;
  const int wv   = threadIdx.x >> 5;
  const int m0 = blockIdx.y * 64 + (wv >> 1) * 16;
  const int n0 = blockIdx.x * 128 + (wv & 1) * 64;
  v8f zero8 = {0.f,0.f,0.f,0.f,0.f,0.f,0.f,0.f};
  v8f acc[4] = {zero8, zero8, zero8, zero8};
  for (int k0 = 0; k0 < K; k0 += 32) {
    v16bf a = load_a_frag(A, m0, K, k0, lane);
#pragma unroll
    for (int j = 0; j < 4; ++j) {
      v16bf b = load_b_frag(W, n0 + j * 16, K, k0, lane);
      acc[j] = __builtin_amdgcn_wmma_f32_16x16x32_bf16(false, a, false, b, (short)0, acc[j], false, false);
    }
  }
  const int nlane = lane & 15;
  const int rh    = (lane >> 4) << 3;   // C/D layout: VGPR v holds row m = v + 8*half
#pragma unroll
  for (int j = 0; j < 4; ++j) {
    float* af = reinterpret_cast<float*>(&acc[j]);
#pragma unroll
    for (int v = 0; v < 8; ++v) {
      int m = m0 + rh + v;
      int n = n0 + j * 16 + nlane;
      float val = af[v] * scale[n] + shift[n];
      if (out_bf) out_bf[(size_t)m * N + n] = f2bf(val);
      else        out_f [(size_t)m * N + n] = val;
    }
  }
}

// ---------- persistent LSTM scan ----------
// 32 workgroups; wg owns 32 h-columns (c0..c0+31). 8 waves: wave wv -> gate g=wv>>1,
// 16-col subtile j=wv&1, both M-tiles (b=0..15, 16..31). Per step:
// z = gates[t] + h_prev * Wh^T  for owned columns, then elementwise LSTM cell.
__global__ __launch_bounds__(256) void lstm_scan(
    const __bf16* __restrict__ gates,  // [T][32][4096] bf16 (BN-folded input gates)
    const __bf16* __restrict__ Wh,     // [4096][1024] bf16
    __bf16* __restrict__ hall,         // [T][32][1024] bf16
    __bf16* __restrict__ hbuf,         // [2][32][1024] bf16 (double buffer, pre-zeroed)
    unsigned* __restrict__ counter) {  // pre-zeroed
  __shared__ float zbuf[4][B2][32];
  const int tid  = threadIdx.x;
  const int lane = tid & 31;
  const int wv   = tid >> 5;
  const int c0   = blockIdx.x * 32;
  const int g    = wv >> 1;
  const int jcol = (wv & 1) * 16;
  const int nrow = g * HID + c0 + jcol;     // Wh row block (gate channel)
  const int eb   = tid >> 3;                 // elementwise: batch row 0..31
  const int ehc  = (tid & 7) * 4;            // elementwise: 4 consecutive h-cols
  float cst[4] = {0.f, 0.f, 0.f, 0.f};
  unsigned target = 0;
  const v8f zero8 = {0.f,0.f,0.f,0.f,0.f,0.f,0.f,0.f};

  for (int t = 0; t < T_STEPS; ++t) {
    const __bf16* hprev = hbuf + (size_t)(t & 1) * B2 * HID;
    __bf16*       hnext = hbuf + (size_t)((t + 1) & 1) * B2 * HID;
    v8f acc0 = zero8, acc1 = zero8;
    for (int k0 = 0; k0 < HID; k0 += 32) {
      v16bf b  = load_b_frag(Wh, nrow, HID, k0, lane);
      v16bf a0 = load_a_frag(hprev, 0,  HID, k0, lane);
      v16bf a1 = load_a_frag(hprev, 16, HID, k0, lane);
      acc0 = __builtin_amdgcn_wmma_f32_16x16x32_bf16(false, a0, false, b, (short)0, acc0, false, false);
      acc1 = __builtin_amdgcn_wmma_f32_16x16x32_bf16(false, a1, false, b, (short)0, acc1, false, false);
    }
    const int colc = jcol + (lane & 15);
    const int rh   = (lane >> 4) << 3;
    float* a0f = reinterpret_cast<float*>(&acc0);
    float* a1f = reinterpret_cast<float*>(&acc1);
#pragma unroll
    for (int v = 0; v < 8; ++v) {
      zbuf[g][rh + v][colc]      = a0f[v];
      zbuf[g][rh + v + 16][colc] = a1f[v];
    }
    __syncthreads();
    const __bf16* gt = gates + (size_t)t * B2 * G4H + (size_t)eb * G4H + c0;
#pragma unroll
    for (int r = 0; r < 4; ++r) {
      int hc = ehc + r;
      float zf = zbuf[0][eb][hc] + bf2f(gt[0 * HID + hc]);
      float zi = zbuf[1][eb][hc] + bf2f(gt[1 * HID + hc]);
      float zo = zbuf[2][eb][hc] + bf2f(gt[2 * HID + hc]);
      float zc = zbuf[3][eb][hc] + bf2f(gt[3 * HID + hc]);
      float fg = 1.f / (1.f + expf(-zf));
      float ig = 1.f / (1.f + expf(-zi));
      float og = 1.f / (1.f + expf(-zo));
      float cc = tanhf(zc);
      float ct = ig * cc + fg * cst[r];
      cst[r] = ct;
      float ht = og * tanhf(ct);
      __bf16 hb = f2bf(ht);
      hnext[(size_t)eb * HID + c0 + hc] = hb;
      hall[((size_t)t * B2 + eb) * HID + c0 + hc] = hb;
    }
    // device-wide barrier (monotonic counter, 32 resident workgroups)
    __threadfence();
    __syncthreads();
    if (tid == 0) {
      __hip_atomic_fetch_add(counter, 1u, __ATOMIC_RELEASE, __HIP_MEMORY_SCOPE_AGENT);
      target += gridDim.x;
      while (__hip_atomic_load(counter, __ATOMIC_ACQUIRE, __HIP_MEMORY_SCOPE_AGENT) < target) {
        __builtin_amdgcn_s_sleep(2);
      }
    }
    __syncthreads();
  }
}

// ---------- small helper kernels ----------
__global__ void cast_f32_bf16_k(const float* __restrict__ in, __bf16* __restrict__ out, int n) {
  int i = blockIdx.x * blockDim.x + threadIdx.x;
  if (i < n) out[i] = f2bf(in[i]);
}

__global__ void bn_fold_k(const float* __restrict__ g, const float* __restrict__ b,
                          const float* __restrict__ m, const float* __restrict__ v,
                          float* __restrict__ scale, float* __restrict__ shift, int n) {
  int i = blockIdx.x * blockDim.x + threadIdx.x;
  if (i < n) {
    float s = g[i] * rsqrtf(v[i] + 1e-5f);
    scale[i] = s;
    shift[i] = b[i] - m[i] * s;
  }
}

// x2 layer0: [T][2B][D], second half of batch is time-reversed x
__global__ void build_x2_l0(const float* __restrict__ x, __bf16* __restrict__ out) {
  int idx = blockIdx.x * blockDim.x + threadIdx.x;   // T*32*512 total
  int d = idx & (DIN - 1); int rest = idx >> 9;
  int b = rest & 31; int t = rest >> 5;
  float v = (b < BATCH) ? x[((size_t)t * BATCH + b) * DIN + d]
                        : x[((size_t)(T_STEPS - 1 - t) * BATCH + (b - BATCH)) * DIN + d];
  out[idx] = f2bf(v);
}

// x2 layer1: [T][2B][2H] built from layer-0 hall [T][32][H]
__global__ void build_x2_l1(const __bf16* __restrict__ hall, __bf16* __restrict__ out) {
  int idx = blockIdx.x * blockDim.x + threadIdx.x;   // T*32*2048 total
  int d = idx & (2 * HID - 1); int rest = idx >> 11;
  int b = rest & 31; int t = rest >> 5;
  __bf16 v;
  if (b < BATCH) {
    v = (d < HID) ? hall[((size_t)t * B2 + b) * HID + d]
                  : hall[((size_t)(T_STEPS - 1 - t) * B2 + BATCH + b) * HID + (d - HID)];
  } else {
    v = (d < HID) ? hall[((size_t)(T_STEPS - 1 - t) * B2 + (b - BATCH)) * HID + d]
                  : hall[((size_t)t * B2 + b) * HID + (d - HID)];
  }
  out[idx] = v;
}

// final X: [T*B][2H] from layer-1 hall
__global__ void build_xfin(const __bf16* __restrict__ hall, __bf16* __restrict__ out) {
  int idx = blockIdx.x * blockDim.x + threadIdx.x;   // T*B*2048 total
  int d = idx & (2 * HID - 1); int rest = idx >> 11;
  int b = rest & 15; int t = rest >> 4;
  __bf16 v = (d < HID) ? hall[((size_t)t * B2 + b) * HID + d]
                       : hall[((size_t)(T_STEPS - 1 - t) * B2 + BATCH + b) * HID + (d - HID)];
  out[idx] = v;
}

__global__ void zero_init(unsigned* __restrict__ hbuf_u32, unsigned* __restrict__ counter) {
  int i = blockIdx.x * blockDim.x + threadIdx.x;
  if (i < 2 * B2 * HID / 2) hbuf_u32[i] = 0;   // 2*32*1024 bf16 = 32768 dwords
  if (i == 0) *counter = 0;
}

__global__ __launch_bounds__(256) void log_softmax_rows(float* __restrict__ out) {
  __shared__ float red[256];
  const int tid = threadIdx.x;
  float* p = out + (size_t)blockIdx.x * OUTF;
  float mx = -3.4e38f;
  for (int j = tid; j < OUTF; j += 256) mx = fmaxf(mx, p[j]);
  red[tid] = mx; __syncthreads();
  for (int s = 128; s > 0; s >>= 1) { if (tid < s) red[tid] = fmaxf(red[tid], red[tid + s]); __syncthreads(); }
  mx = red[0]; __syncthreads();
  float sum = 0.f;
  for (int j = tid; j < OUTF; j += 256) sum += expf(p[j] - mx);
  red[tid] = sum; __syncthreads();
  for (int s = 128; s > 0; s >>= 1) { if (tid < s) red[tid] += red[tid + s]; __syncthreads(); }
  float lse = mx + logf(red[0]);
  for (int j = tid; j < OUTF; j += 256) p[j] -= lse;
}

extern "C" void kernel_launch(void* const* d_in, const int* in_sizes, int n_in,
                              void* d_out, int out_size, void* d_ws, size_t ws_size,
                              hipStream_t stream) {
  const float* x    = (const float*)d_in[0];
  const float* Wx0  = (const float*)d_in[1];
  const float* Wh0  = (const float*)d_in[2];
  const float* g0   = (const float*)d_in[3];
  const float* be0  = (const float*)d_in[4];
  const float* mu0  = (const float*)d_in[5];
  const float* va0  = (const float*)d_in[6];
  const float* Wx1  = (const float*)d_in[7];
  const float* Wh1  = (const float*)d_in[8];
  const float* g1   = (const float*)d_in[9];
  const float* be1  = (const float*)d_in[10];
  const float* mu1  = (const float*)d_in[11];
  const float* va1  = (const float*)d_in[12];
  const float* Wfin = (const float*)d_in[13];
  const float* gf   = (const float*)d_in[14];
  const float* bef  = (const float*)d_in[15];
  const float* muf  = (const float*)d_in[16];
  const float* vaf  = (const float*)d_in[17];
  float* outp = (float*)d_out;

  char* ws = (char*)d_ws;
  size_t off = 0;
  auto alloc = [&](size_t bytes) -> void* {
    void* p = ws + off;
    off = (off + bytes + 255) & ~(size_t)255;
    return p;
  };
  __bf16* wx0b  = (__bf16*)alloc((size_t)G4H * DIN * 2);
  __bf16* wh0b  = (__bf16*)alloc((size_t)G4H * HID * 2);
  __bf16* wx1b  = (__bf16*)alloc((size_t)G4H * 2 * HID * 2);
  __bf16* wh1b  = (__bf16*)alloc((size_t)G4H * HID * 2);
  __bf16* wfinb = (__bf16*)alloc((size_t)OUTF * 2 * HID * 2);
  float* sc0 = (float*)alloc(G4H * 4);  float* sh0 = (float*)alloc(G4H * 4);
  float* sc1 = (float*)alloc(G4H * 4);  float* sh1 = (float*)alloc(G4H * 4);
  float* scf = (float*)alloc(OUTF * 4); float* shf = (float*)alloc(OUTF * 4);
  __bf16* x2     = (__bf16*)alloc((size_t)T_STEPS * B2 * 2 * HID * 2);   // reused for XFIN
  __bf16* gatesb = (__bf16*)alloc((size_t)T_STEPS * B2 * G4H * 2);
  __bf16* hall   = (__bf16*)alloc((size_t)T_STEPS * B2 * HID * 2);
  __bf16* hbuf   = (__bf16*)alloc((size_t)2 * B2 * HID * 2);
  unsigned* cnt  = (unsigned*)alloc(256);

  const int thr = 256;
  // weight casts to bf16
  cast_f32_bf16_k<<<(G4H * DIN) / thr, thr, 0, stream>>>(Wx0, wx0b, G4H * DIN);
  cast_f32_bf16_k<<<(G4H * HID) / thr, thr, 0, stream>>>(Wh0, wh0b, G4H * HID);
  cast_f32_bf16_k<<<(G4H * 2 * HID) / thr, thr, 0, stream>>>(Wx1, wx1b, G4H * 2 * HID);
  cast_f32_bf16_k<<<(G4H * HID) / thr, thr, 0, stream>>>(Wh1, wh1b, G4H * HID);
  cast_f32_bf16_k<<<(OUTF * 2 * HID) / thr, thr, 0, stream>>>(Wfin, wfinb, OUTF * 2 * HID);
  // BN folds
  bn_fold_k<<<G4H / thr, thr, 0, stream>>>(g0, be0, mu0, va0, sc0, sh0, G4H);
  bn_fold_k<<<G4H / thr, thr, 0, stream>>>(g1, be1, mu1, va1, sc1, sh1, G4H);
  bn_fold_k<<<OUTF / thr, thr, 0, stream>>>(gf, bef, muf, vaf, scf, shf, OUTF);

  // ---- layer 0 ----
  build_x2_l0<<<(T_STEPS * B2 * DIN) / thr, thr, 0, stream>>>(x, x2);
  gemm_bf16_epi<<<dim3(G4H / 128, (T_STEPS * B2) / 64), thr, 0, stream>>>(
      x2, wx0b, sc0, sh0, gatesb, nullptr, G4H, DIN);
  zero_init<<<128, thr, 0, stream>>>((unsigned*)hbuf, cnt);
  lstm_scan<<<32, thr, 0, stream>>>(gatesb, wh0b, hall, hbuf, cnt);

  // ---- layer 1 ----
  build_x2_l1<<<(T_STEPS * B2 * 2 * HID) / thr, thr, 0, stream>>>(hall, x2);
  gemm_bf16_epi<<<dim3(G4H / 128, (T_STEPS * B2) / 64), thr, 0, stream>>>(
      x2, wx1b, sc1, sh1, gatesb, nullptr, G4H, 2 * HID);
  zero_init<<<128, thr, 0, stream>>>((unsigned*)hbuf, cnt);
  lstm_scan<<<32, thr, 0, stream>>>(gatesb, wh1b, hall, hbuf, cnt);

  // ---- final projection + BN + log_softmax ----
  build_xfin<<<(T_STEPS * BATCH * 2 * HID) / thr, thr, 0, stream>>>(hall, x2);
  gemm_bf16_epi<<<dim3(OUTF / 128, (T_STEPS * BATCH) / 64), thr, 0, stream>>>(
      x2, wfinb, scf, shf, nullptr, outp, OUTF, 2 * HID);
  log_softmax_rows<<<T_STEPS * BATCH, thr, 0, stream>>>(outp);
}
